// Linears_25950192403135
// MI455X (gfx1250) — compile-verified, matrix operand order checked by reading, not run on hardware
//
#include <hip/hip_runtime.h>

#define D_DIM 1024
#define BATCH 4
#define LSEQ  192
#define MROWS 768   // BATCH * LSEQ

typedef __attribute__((ext_vector_type(16))) __bf16 v16bf;
typedef __attribute__((ext_vector_type(8)))  float  v8f;

union FragBF { unsigned u[8]; v16bf v; };
union FU     { float f; unsigned u; };

__device__ __forceinline__ unsigned short f2bf(float x) {
    FU t; t.f = x;
    unsigned r = t.u + 0x7FFFu + ((t.u >> 16) & 1u);   // round-to-nearest-even
    return (unsigned short)(r >> 16);
}
__device__ __forceinline__ float bf2f(unsigned short h) {
    FU t; t.u = ((unsigned)h) << 16;
    return t.f;
}

__device__ __forceinline__ float fast_tanh(float x) {
#if __has_builtin(__builtin_amdgcn_tanhf)
    return __builtin_amdgcn_tanhf(x);
#elif __has_builtin(__builtin_amdgcn_exp2f) && __has_builtin(__builtin_amdgcn_rcpf)
    // tanh(x) = 1 - 2/(exp2(2x*log2e)+1) ; saturates correctly at +/-inf
    float e = __builtin_amdgcn_exp2f(x * 2.8853900817779268f);
    return 1.0f - 2.0f * __builtin_amdgcn_rcpf(e + 1.0f);
#else
    return tanhf(x);
#endif
}

// C[768,1024] = A[768,1024] x W[1024,1024] + bias, via bf16 hi/lo split WMMA
// (3 wmma per tile-pair: hi*hi + hi*lo + lo*hi -> ~2^-16 relative error).
// Block: 256 threads (8 waves), tile 128(M) x 64(N), K-chunk 32.
__global__ __launch_bounds__(256)
void gemm_wmma_bf16x2(const float* __restrict__ A, const float* __restrict__ W,
                      const float* __restrict__ bias, float* __restrict__ C) {
    __shared__ unsigned       shAh[128 * 17];   // A hi, packed bf16 pairs, row stride 17 dw
    __shared__ unsigned       shAl[128 * 17];   // A lo
    __shared__ unsigned short shBh[64 * 34];    // W hi, column-major [n][k], 34 shorts/row
    __shared__ unsigned short shBl[64 * 34];    // W lo

    const int tid  = threadIdx.x;
    const int m0   = blockIdx.x * 128;
    const int n0   = blockIdx.y * 64;
    const int wave = tid >> 5;
    const int lane = tid & 31;
    const int lh   = lane >> 4;      // lane half (0/1)
    const int lr   = lane & 15;      // row/col within 16
    const int wr   = wave >> 1;      // 0..3 -> 32-row band
    const int wc   = wave & 1;       // 0..1 -> 32-col band

    v8f acc[2][2] = {};

    for (int kc = 0; kc < D_DIM; kc += 32) {
        // ---- stage A tile: 128 rows x 32 k, f32 -> bf16 hi/lo, packed pairs ----
        {
            int r   = tid >> 1;
            int kp0 = (tid & 1) * 8;                       // bf16-pair index base
            const float* src = A + (m0 + r) * D_DIM + kc + kp0 * 2;
            #pragma unroll
            for (int p = 0; p < 8; ++p) {
                float x0 = src[2 * p], x1 = src[2 * p + 1];
                unsigned short h0 = f2bf(x0), h1 = f2bf(x1);
                float l0 = x0 - bf2f(h0),  l1 = x1 - bf2f(h1);
                shAh[r * 17 + kp0 + p] = (unsigned)h0 | ((unsigned)h1 << 16);
                shAl[r * 17 + kp0 + p] = (unsigned)f2bf(l0) | ((unsigned)f2bf(l1) << 16);
            }
        }
        // ---- stage W tile: 32 k x 64 n, stored column-major so K-pairs are contiguous ----
        {
            int kk = tid >> 3;            // 0..31
            int nn = (tid & 7) * 8;       // 0..56
            const float* src = W + (kc + kk) * D_DIM + n0 + nn;
            #pragma unroll
            for (int e = 0; e < 8; ++e) {
                float x = src[e];
                unsigned short h = f2bf(x);
                shBh[(nn + e) * 34 + kk] = h;
                shBl[(nn + e) * 34 + kk] = f2bf(x - bf2f(h));
            }
        }
        __syncthreads();

        const unsigned* Bh32 = (const unsigned*)shBh;   // 17 dwords per column row
        const unsigned* Bl32 = (const unsigned*)shBl;

        FragBF ah[2], al[2], bh[2], bl[2];
        #pragma unroll
        for (int rt = 0; rt < 2; ++rt) {
            int row = wr * 32 + rt * 16 + lr;
            #pragma unroll
            for (int v = 0; v < 8; ++v) {
                int kp = (v < 4 ? v : v + 4) + 4 * lh;   // ISA 16-bit A-fragment K map
                ah[rt].u[v] = shAh[row * 17 + kp];
                al[rt].u[v] = shAl[row * 17 + kp];
            }
        }
        #pragma unroll
        for (int ct = 0; ct < 2; ++ct) {
            int col = wc * 32 + ct * 16 + lr;
            #pragma unroll
            for (int v = 0; v < 8; ++v) {
                int kp = (v < 4 ? v : v + 4) + 4 * lh;
                bh[ct].u[v] = Bh32[col * 17 + kp];
                bl[ct].u[v] = Bl32[col * 17 + kp];
            }
        }

        #pragma unroll
        for (int rt = 0; rt < 2; ++rt)
            #pragma unroll
            for (int ct = 0; ct < 2; ++ct) {
                acc[rt][ct] = __builtin_amdgcn_wmma_f32_16x16x32_bf16(
                    false, ah[rt].v, false, bh[ct].v, (short)0, acc[rt][ct], false, false);
                acc[rt][ct] = __builtin_amdgcn_wmma_f32_16x16x32_bf16(
                    false, ah[rt].v, false, bl[ct].v, (short)0, acc[rt][ct], false, false);
                acc[rt][ct] = __builtin_amdgcn_wmma_f32_16x16x32_bf16(
                    false, al[rt].v, false, bh[ct].v, (short)0, acc[rt][ct], false, false);
            }
        __syncthreads();
    }

    // Epilogue: C/D layout — lane lr = N, dword v -> M = v + 8*lh
    #pragma unroll
    for (int rt = 0; rt < 2; ++rt) {
        #pragma unroll
        for (int ct = 0; ct < 2; ++ct) {
            int n = n0 + wc * 32 + ct * 16 + lr;
            float bn = bias[n];
            #pragma unroll
            for (int v = 0; v < 8; ++v) {
                int m = m0 + wr * 32 + rt * 16 + (v + 8 * lh);
                C[m * D_DIM + n] = acc[rt][ct][v] + bn;
            }
        }
    }
}

// out[b,i,j] = b3 + sum_d tanh(q[b,i,d]+k[b,j,d]) * w3[d]
// 16x16 (i,j) tile per block; q/k staged through LDS in 256-wide d-chunks;
// row stride 257 (=1 mod 64) -> conflict-free k-row reads across the 16 j lanes.
__global__ __launch_bounds__(256)
void tanh_score_kernel(const float* __restrict__ q, const float* __restrict__ kbuf,
                       const float* __restrict__ w3, const float* __restrict__ b3,
                       float* __restrict__ out) {
    __shared__ float qs[16 * 257];
    __shared__ float ks[16 * 257];
    __shared__ float ws[256];

    const int tid = threadIdx.x;
    const int bid = blockIdx.x;
    const int b   = bid / 144;
    const int ti  = (bid % 144) / 12;
    const int tj  = bid % 12;
    const int i0  = ti * 16, j0 = tj * 16;

    const int i = tid >> 4, j = tid & 15;
    float acc = 0.0f;

    for (int d0 = 0; d0 < D_DIM; d0 += 256) {
        #pragma unroll
        for (int s = 0; s < 16; ++s) {       // coalesced: all 256 lanes load one row
            qs[s * 257 + tid] = q[(b * LSEQ + i0 + s) * D_DIM + d0 + tid];
            ks[s * 257 + tid] = kbuf[(b * LSEQ + j0 + s) * D_DIM + d0 + tid];
        }
        ws[tid] = w3[d0 + tid];
        __syncthreads();

        const float* qrow = &qs[i * 257];
        const float* krow = &ks[j * 257];
        #pragma unroll 8
        for (int d = 0; d < 256; ++d) {
            acc = fmaf(fast_tanh(qrow[d] + krow[d]), ws[d], acc);
        }
        __syncthreads();
    }
    out[(b * LSEQ + i0 + i) * LSEQ + j0 + j] = acc + b3[0];
}

extern "C" void kernel_launch(void* const* d_in, const int* in_sizes, int n_in,
                              void* d_out, int out_size, void* d_ws, size_t ws_size,
                              hipStream_t stream) {
    const float* hs  = (const float*)d_in[0];
    const float* ehs = (const float*)d_in[1];
    const float* W1  = (const float*)d_in[2];
    const float* b1  = (const float*)d_in[3];
    const float* W2  = (const float*)d_in[4];
    const float* b2  = (const float*)d_in[5];
    const float* W3  = (const float*)d_in[6];
    const float* b3  = (const float*)d_in[7];
    float* outp = (float*)d_out;

    float* qbuf = (float*)d_ws;                 // [768,1024] f32  (3 MB)
    float* kbuf = qbuf + (size_t)MROWS * D_DIM; // [768,1024] f32  (3 MB)

    dim3 grid(MROWS / 128, D_DIM / 64), blk(256);
    hipLaunchKernelGGL(gemm_wmma_bf16x2, grid, blk, 0, stream, hs,  W1, b1, qbuf);
    hipLaunchKernelGGL(gemm_wmma_bf16x2, grid, blk, 0, stream, ehs, W2, b2, kbuf);
    hipLaunchKernelGGL(tanh_score_kernel, dim3(BATCH * 12 * 12), blk, 0, stream,
                       qbuf, kbuf, W3, b3, outp);
}
